// Linear_27462020890731
// MI455X (gfx1250) — matrix operand order repaired and checked by reading, not compiled, not run on hardware
//
#include <hip/hip_runtime.h>

// ---------------------------------------------------------------------------
// LoRA-fused Linear for MI455X (gfx1250, wave32, WMMA).
// out = X @ W^T + b + (X_k @ A_k^T) @ B_k^T      (SCALING = alpha/r = 1.0)
//
// fp32 inputs are split into bf16 hi/lo pairs; base GEMM = 3 bf16 WMMAs
// (hi*hi + hi*lo + lo*hi) per 16x16x32 tile -> ~fp32 input fidelity at the
// bf16 XDL rate. If d_ws is large enough, X and W are pre-split once into
// packed bf16 arrays so the GEMM inner loop is pure loads + v_wmma (no
// conversion VALU); otherwise falls back to converting on the fly.
// Rank-16 LoRA delta is a fp32-FMA epilogue over precomputed h = X @ A^T.
// ---------------------------------------------------------------------------

typedef __attribute__((ext_vector_type(16))) __bf16 v16bf;
typedef __attribute__((ext_vector_type(8)))  __bf16 v8bf;
typedef __attribute__((ext_vector_type(8)))  float  v8f;
typedef __attribute__((ext_vector_type(4)))  float  v4f;

#define DIN   4096
#define DOUT  4096
#define RNK   16
#define MTOT  (32 * 512)      // 16384 rows total
#define GRPSH 10              // rows per LoRA group = 2*512 = 1024 = 1<<10

#define WELEMS ((size_t)DOUT * DIN)   // 16,777,216
#define XELEMS ((size_t)MTOT * DIN)   // 67,108,864

// ---------------------------------------------------------------------------
// Kernel 1: h[m, r] = dot(x[m, :], lora_a[m>>10, r, :])   (fp32, memory-bound)
// ---------------------------------------------------------------------------
__global__ __launch_bounds__(256) void lora_h_kernel(
    const float* __restrict__ x,
    const float* __restrict__ lora_a,
    float* __restrict__ h) {
  const int m   = blockIdx.x;
  const int tid = threadIdx.x;
  const int r   = tid >> 4;
  const int s   = tid & 15;

  const float* xr = x + (size_t)m * DIN;
  const float* ar = lora_a + ((size_t)(m >> GRPSH) * RNK + r) * DIN;

  float acc = 0.0f;
  for (int i = s; i < DIN; i += 16)
    acc = fmaf(xr[i], ar[i], acc);

  acc += __shfl_xor(acc, 8, 16);
  acc += __shfl_xor(acc, 4, 16);
  acc += __shfl_xor(acc, 2, 16);
  acc += __shfl_xor(acc, 1, 16);
  if (s == 0) h[(size_t)m * RNK + r] = acc;
}

// ---------------------------------------------------------------------------
// Kernel 1b: split a fp32 array into bf16 hi/lo arrays (bandwidth-bound).
// ---------------------------------------------------------------------------
__global__ __launch_bounds__(256) void pack_hilo_kernel(
    const float* __restrict__ src,
    __bf16* __restrict__ hi,
    __bf16* __restrict__ lo,
    unsigned long long n) {
  const size_t idx = ((size_t)blockIdx.x * 256 + threadIdx.x) * 4;
  if (idx >= n) return;
  v4f v = *(const v4f*)(src + idx);
  v4f vh, vl;
  __bf16 hb[4], lb[4];
#pragma unroll
  for (int e = 0; e < 4; ++e) {
    hb[e] = (__bf16)v[e];                  // RNE
    lb[e] = (__bf16)(v[e] - (float)hb[e]); // residual
  }
  typedef __attribute__((ext_vector_type(4))) __bf16 v4bf;
  v4bf ho = {hb[0], hb[1], hb[2], hb[3]};
  v4bf lh = {lb[0], lb[1], lb[2], lb[3]};
  *(v4bf*)(hi + idx) = ho;
  *(v4bf*)(lo + idx) = lh;
}

// ---------------------------------------------------------------------------
// Fragment loaders.
// A (16x32, MxK): lane L holds row M=L&15; K-chunks {0..7,16..23} for L<16,
//   {8..15,24..31} for L>=16 (ISA 7.12.2, 16-bit A 16x32).
// B (32x16, KxN): lane L holds column N=L&15 (= contiguous row N of W since
//   B = W^T), K-chunks {0..7,16..23} for L<16, {8..15,24..31} for L>=16.
// ---------------------------------------------------------------------------
__device__ __forceinline__ void split4(const v4f c, int base, v16bf& hi, v16bf& lo) {
#pragma unroll
  for (int e = 0; e < 4; ++e) {
    const float f  = c[e];
    const __bf16 h = (__bf16)f;
    hi[base + e] = h;
    lo[base + e] = (__bf16)(f - (float)h);
  }
}

__device__ __forceinline__ void load_a_frag_f32(const float* __restrict__ X,
                                                int row0, int k0, int lane,
                                                v16bf& hi, v16bf& lo) {
  const int half = lane >> 4;
  const int mr   = lane & 15;
  const float* p = X + (size_t)(row0 + mr) * DIN + k0 + half * 8;
  split4(*(const v4f*)(p),      0, hi, lo);
  split4(*(const v4f*)(p + 4),  4, hi, lo);
  split4(*(const v4f*)(p + 16), 8, hi, lo);
  split4(*(const v4f*)(p + 20), 12, hi, lo);
}

__device__ __forceinline__ void load_b_frag_f32(const float* __restrict__ W,
                                                int col0, int k0, int lane,
                                                v16bf& hi, v16bf& lo) {
  const int nc   = lane & 15;
  const int kh   = lane >> 4;
  const float* p = W + (size_t)(col0 + nc) * DIN + k0 + kh * 8;
  split4(*(const v4f*)(p),      0, hi, lo);
  split4(*(const v4f*)(p + 4),  4, hi, lo);
  split4(*(const v4f*)(p + 16), 8, hi, lo);
  split4(*(const v4f*)(p + 20), 12, hi, lo);
}

__device__ __forceinline__ v16bf combine8(v8bf c0, v8bf c1) {
  v16bf f;
#pragma unroll
  for (int e = 0; e < 8; ++e) { f[e] = c0[e]; f[8 + e] = c1[e]; }
  return f;
}

__device__ __forceinline__ v16bf load_a_frag_bf(const __bf16* __restrict__ P,
                                                int row0, int k0, int lane) {
  const int half = lane >> 4;
  const int mr   = lane & 15;
  const __bf16* p = P + (size_t)(row0 + mr) * DIN + k0 + half * 8;
  return combine8(*(const v8bf*)(p), *(const v8bf*)(p + 16));
}

__device__ __forceinline__ v16bf load_b_frag_bf(const __bf16* __restrict__ P,
                                                int col0, int k0, int lane) {
  const int nc = lane & 15;
  const int kh = lane >> 4;
  const __bf16* p = P + (size_t)(col0 + nc) * DIN + k0 + kh * 8;
  return combine8(*(const v8bf*)(p), *(const v8bf*)(p + 16));
}

// ---------------------------------------------------------------------------
// Kernel 2: main GEMM + bias + LoRA epilogue.
// Workgroup tile 128M x 128N; 8 waves, each 32M x 64N (2x4 WMMA tiles).
// ---------------------------------------------------------------------------
template <bool PACKED>
__global__ __launch_bounds__(256) void lora_gemm_kernel(
    const float* __restrict__ X,
    const float* __restrict__ W,
    const __bf16* __restrict__ Xhi, const __bf16* __restrict__ Xlo,
    const __bf16* __restrict__ Whi, const __bf16* __restrict__ Wlo,
    const float* __restrict__ bias,
    const float* __restrict__ h,
    const float* __restrict__ lora_b,
    float* __restrict__ out) {
  const int tid  = threadIdx.x;
  const int lane = tid & 31;
  const int wave = tid >> 5;
  const int wm   = wave & 3;
  const int wn   = wave >> 2;

  const int tile_n0 = blockIdx.x * 128 + wn * 64;
  const int tile_m0 = blockIdx.y * 128 + wm * 32;

  v8f acc[2][4];
#pragma unroll
  for (int i = 0; i < 2; ++i)
#pragma unroll
    for (int j = 0; j < 4; ++j) acc[i][j] = (v8f)0.0f;

  for (int k0 = 0; k0 < DIN; k0 += 32) {
    v16bf ahi[2], alo[2], bhi[4], blo[4];
    if constexpr (PACKED) {
#pragma unroll
      for (int i = 0; i < 2; ++i) {
        ahi[i] = load_a_frag_bf(Xhi, tile_m0 + 16 * i, k0, lane);
        alo[i] = load_a_frag_bf(Xlo, tile_m0 + 16 * i, k0, lane);
      }
#pragma unroll
      for (int j = 0; j < 4; ++j) {
        bhi[j] = load_b_frag_bf(Whi, tile_n0 + 16 * j, k0, lane);
        blo[j] = load_b_frag_bf(Wlo, tile_n0 + 16 * j, k0, lane);
      }
    } else {
#pragma unroll
      for (int i = 0; i < 2; ++i)
        load_a_frag_f32(X, tile_m0 + 16 * i, k0, lane, ahi[i], alo[i]);
#pragma unroll
      for (int j = 0; j < 4; ++j)
        load_b_frag_f32(W, tile_n0 + 16 * j, k0, lane, bhi[j], blo[j]);
    }

#pragma unroll
    for (int i = 0; i < 2; ++i) {
#pragma unroll
      for (int j = 0; j < 4; ++j) {
        acc[i][j] = __builtin_amdgcn_wmma_f32_16x16x32_bf16(
            false, ahi[i], false, bhi[j], (short)0, acc[i][j], false, false);
        acc[i][j] = __builtin_amdgcn_wmma_f32_16x16x32_bf16(
            false, ahi[i], false, blo[j], (short)0, acc[i][j], false, false);
        acc[i][j] = __builtin_amdgcn_wmma_f32_16x16x32_bf16(
            false, alo[i], false, bhi[j], (short)0, acc[i][j], false, false);
      }
    }
  }

  // ---- epilogue: + bias + rank-16 LoRA delta (SCALING == 1.0) ----
  // C/D layout (ISA 7.12.2): VGPR v holds M = v + 8*(lane>=16), N = lane&15.
  const int half = lane >> 4;
  const int nc   = lane & 15;
  const int kgrp = tile_m0 >> GRPSH;  // 32-row wave tile never crosses a group

  float lb[4][RNK];
  float bb[4];
#pragma unroll
  for (int j = 0; j < 4; ++j) {
    const int col = tile_n0 + 16 * j + nc;
    bb[j] = bias[col];
    const float* lbp = lora_b + ((size_t)kgrp * DOUT + col) * RNK;
#pragma unroll
    for (int r = 0; r < RNK; ++r) lb[j][r] = lbp[r];
  }

#pragma unroll
  for (int i = 0; i < 2; ++i) {
#pragma unroll
    for (int v = 0; v < 8; ++v) {
      const int row = tile_m0 + 16 * i + v + 8 * half;
      const float* hr = h + (size_t)row * RNK;
      float hv[RNK];
#pragma unroll
      for (int r = 0; r < RNK; ++r) hv[r] = hr[r];
#pragma unroll
      for (int j = 0; j < 4; ++j) {
        const int col = tile_n0 + 16 * j + nc;
        float d = acc[i][j][v] + bb[j];
#pragma unroll
        for (int r = 0; r < RNK; ++r) d = fmaf(hv[r], lb[j][r], d);
        out[(size_t)row * DOUT + col] = d;
      }
    }
  }
}

// ---------------------------------------------------------------------------
// Launch
// ---------------------------------------------------------------------------
extern "C" void kernel_launch(void* const* d_in, const int* in_sizes, int n_in,
                              void* d_out, int out_size, void* d_ws, size_t ws_size,
                              hipStream_t stream) {
  const float* x      = (const float*)d_in[0];  // [32, 512, 4096]
  const float* base_w = (const float*)d_in[1];  // [4096, 4096]
  const float* base_b = (const float*)d_in[2];  // [4096]
  const float* lora_a = (const float*)d_in[3];  // [16, 16, 4096]
  const float* lora_b = (const float*)d_in[4];  // [16, 4096, 16]
  float* out = (float*)d_out;                   // [32, 512, 4096]

  // workspace layout
  char* ws = (char*)d_ws;
  const size_t h_bytes  = (size_t)MTOT * RNK * sizeof(float);   //   1 MB
  const size_t w_bytes  = WELEMS * sizeof(__bf16);              //  32 MB each
  const size_t x_bytes  = XELEMS * sizeof(__bf16);              // 128 MB each
  const size_t need_all = h_bytes + 2 * w_bytes + 2 * x_bytes;  // ~321 MB

  float* h = (float*)ws;

  lora_h_kernel<<<MTOT, 256, 0, stream>>>(x, lora_a, h);

  const dim3 gemm_grid(DOUT / 128, MTOT / 128);

  if (ws_size >= need_all) {
    __bf16* Whi = (__bf16*)(ws + h_bytes);
    __bf16* Wlo = (__bf16*)(ws + h_bytes + w_bytes);
    __bf16* Xhi = (__bf16*)(ws + h_bytes + 2 * w_bytes);
    __bf16* Xlo = (__bf16*)(ws + h_bytes + 2 * w_bytes + x_bytes);

    pack_hilo_kernel<<<(unsigned)(WELEMS / (4 * 256)), 256, 0, stream>>>(
        base_w, Whi, Wlo, (unsigned long long)WELEMS);
    pack_hilo_kernel<<<(unsigned)(XELEMS / (4 * 256)), 256, 0, stream>>>(
        x, Xhi, Xlo, (unsigned long long)XELEMS);

    lora_gemm_kernel<true><<<gemm_grid, 256, 0, stream>>>(
        x, base_w, Xhi, Xlo, Whi, Wlo, base_b, h, lora_b, out);
  } else {
    lora_gemm_kernel<false><<<gemm_grid, 256, 0, stream>>>(
        x, base_w, nullptr, nullptr, nullptr, nullptr, base_b, h, lora_b, out);
  }
}
